// SparseMaskFCPredictor_37005438222837
// MI455X (gfx1250) — compile-verified, matrix operand order checked by reading, not run on hardware
//
#include <hip/hip_runtime.h>
#include <stdint.h>

// ---------------- problem constants ----------------
#define NUM_ANCHORS 9
#define KDIM        2304            // C_IN*3*3
#define C_OUT       784
#define NROWS       4096
#define ROW_PAIRS   137             // 32-row padded groups: <= 4096 + 9*31 rounded to 32
#define CAP         (ROW_PAIRS*32)  // 4384 padded rows
#define NUM_TILES   (CAP/16)        // 274 row tiles of 16
#define NTILES      49              // 784/16
#define NPAIRS      25              // ceil(49/2) n-tile pairs

typedef __attribute__((ext_vector_type(16))) _Float16 v16h;
typedef __attribute__((ext_vector_type(8)))  float    v8f;

// ---------------- phase 1: routing / stable counting sort ----------------
// One block, 256 threads, each owning 16 consecutive rows (stable order).
// Expert segments are padded to 32 rows so a 2-row-tile GEMM group never
// straddles an expert boundary.
__global__ __launch_bounds__(256)
void routing_kernel(const int* __restrict__ aidx,
                    int* __restrict__ perm, int* __restrict__ sortpos,
                    int* __restrict__ srcrow, int* __restrict__ dstrow,
                    int* __restrict__ tileExpert) {
  __shared__ int lc[256][NUM_ANCHORS];     // per-thread per-bin counts -> offsets
  __shared__ int gstart[NUM_ANCHORS + 1];  // unpadded segment starts
  __shared__ int gpad[NUM_ANCHORS + 1];    // 32-aligned padded segment starts
  const int t = threadIdx.x;

  int c[NUM_ANCHORS];
#pragma unroll
  for (int b = 0; b < NUM_ANCHORS; ++b) c[b] = 0;
  int bins[16];
#pragma unroll
  for (int r = 0; r < 16; ++r) {
    int a = aidx[t * 16 + r];
    a = a < 0 ? 0 : (a > NUM_ANCHORS - 1 ? NUM_ANCHORS - 1 : a);
    bins[r] = a;
    c[a]++;
  }
#pragma unroll
  for (int b = 0; b < NUM_ANCHORS; ++b) lc[t][b] = c[b];
  __syncthreads();

  if (t == 0) {
    int acc = 0;
    for (int b = 0; b < NUM_ANCHORS; ++b) {       // bin-major, thread-minor scan
      gstart[b] = acc;
      int s = acc;
      for (int tt = 0; tt < 256; ++tt) { int v = lc[tt][b]; lc[tt][b] = s; s += v; }
      acc = s;
    }
    gstart[NUM_ANCHORS] = acc;                    // == NROWS
    int pacc = 0;
    for (int b = 0; b < NUM_ANCHORS; ++b) {
      gpad[b] = pacc;
      int cb = gstart[b + 1] - gstart[b];
      pacc += ((cb + 31) >> 5) << 5;              // pad each expert to 32 rows
    }
    gpad[NUM_ANCHORS] = pacc;
  }
  __syncthreads();

  int off[NUM_ANCHORS];
#pragma unroll
  for (int b = 0; b < NUM_ANCHORS; ++b) off[b] = lc[t][b];
#pragma unroll
  for (int r = 0; r < 16; ++r) {
    int i = t * 16 + r;
    int b = bins[r];
    int pos = off[b]++;
    perm[pos] = i;       // perm[k] = original row at sort position k
    sortpos[i] = pos;    // inverse permutation
  }
  __syncthreads();

  // padded layout: srcrow (gather source), dstrow (routed output row), tile expert
  for (int p = t; p < CAP; p += 256) {
    int e = NUM_ANCHORS - 1;
#pragma unroll
    for (int b = 0; b < NUM_ANCHORS; ++b) {
      if (p < gpad[b + 1]) { e = b; break; }
    }
    int sr = -1, dr = -1;
    if (p < gpad[NUM_ANCHORS]) {
      int local = p - gpad[e];
      int cb = gstart[e + 1] - gstart[e];
      if (local < cb) {
        int k = gstart[e] + local;     // sort position
        sr = perm[k];                  // read row perm[k]
        dr = sortpos[k];               // out[sortpos[k]] = ysorted[k]  (perm[perm] trick)
      }
    }
    srcrow[p] = sr;
    dstrow[p] = dr;
    if ((p & 15) == 0) tileExpert[p >> 4] = e;
  }
}

// ---------------- phase 2a: weights fp32 -> f16 ----------------
__global__ __launch_bounds__(256)
void cvtW_kernel(const float* __restrict__ W, _Float16* __restrict__ Wh, int n) {
  int i = blockIdx.x * 256 + threadIdx.x;
  if (i < n) Wh[i] = (_Float16)W[i];
}

// ---------------- phase 2b: gather + convert x into padded sorted order ----------------
__global__ __launch_bounds__(256)
void gatherX_kernel(const float* __restrict__ x, const int* __restrict__ srcrow,
                    _Float16* __restrict__ Xs) {
  int p = blockIdx.x;
  int sr = srcrow[p];
  _Float16* dst = Xs + (size_t)p * KDIM;
  if (sr < 0) {
    for (int d = threadIdx.x; d < KDIM; d += 256) dst[d] = (_Float16)0.0f;
  } else {
    const float* s = x + (size_t)sr * KDIM;
    for (int d = threadIdx.x; d < KDIM; d += 256) dst[d] = (_Float16)s[d];
  }
}

// ---------------- phase 3: WMMA GEMM + routed scatter ----------------
// grid = (ROW_PAIRS, ceil(NPAIRS/4)), block = 128 (4 waves).
// Each wave: 2 row-tiles x 2 n-tiles = 4 independent accumulators; fragments
// double-buffered so 8 B128 loads stay in flight under the 4 WMMAs.
union Frag { v16h v; uint4 q[2]; };

__device__ __forceinline__ void load_frag(Frag& f, const _Float16* p, int kk) {
  f.q[0] = *(const uint4*)(p + kk);
  f.q[1] = *(const uint4*)(p + kk + 16);
}

__global__ __launch_bounds__(128)
void moe_gemm_kernel(const _Float16* __restrict__ Xs, const _Float16* __restrict__ Wh,
                     const float* __restrict__ bias, const int* __restrict__ tileExpert,
                     const int* __restrict__ dstrow, float* __restrict__ out) {
  const int bx   = blockIdx.x;                 // row-tile pair (32 rows, one expert)
  const int wave = threadIdx.x >> 5;
  const int lane = threadIdx.x & 31;
  const int npair = blockIdx.y * 4 + wave;
  if (npair >= NPAIRS) return;                 // wave-uniform: EXEC all-1s in live waves

  const int rt0 = bx * 2;                      // 16-row tiles rt0, rt0+1: same expert
  const int e   = tileExpert[rt0];
  const int nt1_live = (npair * 2 + 1) < NTILES;

  // 16-bit A-matrix layout: lanes 0-15 -> K 0..7,16..23 ; lanes 16-31 -> K 8..15,24..31
  const int m  = lane & 15;
  const int kb = (lane >> 4) * 8;              // per-lane K base (halves)
  const _Float16* aRow0 = Xs + ((size_t)rt0 * 16 + m) * KDIM + kb;
  const _Float16* aRow1 = aRow0 + (size_t)16 * KDIM;
  const int ncol0 = npair * 32 + (lane & 15);  // B column n == contiguous W row n
  const _Float16* bRow0 = Wh + (size_t)e * C_OUT * KDIM + (size_t)ncol0 * KDIM + kb;
  const _Float16* bRow1 = bRow0 + (size_t)16 * KDIM;   // may read into slack if nt1 dead

  v8f acc00 = {}, acc01 = {}, acc10 = {}, acc11 = {};
  Frag a0c, a1c, b0c, b1c, a0n, a1n, b0n, b1n;

  load_frag(a0c, aRow0, 0);
  load_frag(a1c, aRow1, 0);
  load_frag(b0c, bRow0, 0);
  load_frag(b1c, bRow1, 0);

  for (int kk = 32; kk < KDIM; kk += 32) {
    // issue next iteration's 8 B128 loads before consuming current fragments
    load_frag(a0n, aRow0, kk);
    load_frag(a1n, aRow1, kk);
    load_frag(b0n, bRow0, kk);
    load_frag(b1n, bRow1, kk);
    __builtin_prefetch(bRow0 + kk + 512, 0, 1);   // global_prefetch_b8 (unconditional)
    __builtin_prefetch(bRow1 + kk + 512, 0, 1);

    acc00 = __builtin_amdgcn_wmma_f32_16x16x32_f16(false, a0c.v, false, b0c.v, (short)0, acc00, false, false);
    acc01 = __builtin_amdgcn_wmma_f32_16x16x32_f16(false, a0c.v, false, b1c.v, (short)0, acc01, false, false);
    acc10 = __builtin_amdgcn_wmma_f32_16x16x32_f16(false, a1c.v, false, b0c.v, (short)0, acc10, false, false);
    acc11 = __builtin_amdgcn_wmma_f32_16x16x32_f16(false, a1c.v, false, b1c.v, (short)0, acc11, false, false);

    a0c = a0n; a1c = a1n; b0c = b0n; b1c = b1n;
  }
  acc00 = __builtin_amdgcn_wmma_f32_16x16x32_f16(false, a0c.v, false, b0c.v, (short)0, acc00, false, false);
  acc01 = __builtin_amdgcn_wmma_f32_16x16x32_f16(false, a0c.v, false, b1c.v, (short)0, acc01, false, false);
  acc10 = __builtin_amdgcn_wmma_f32_16x16x32_f16(false, a1c.v, false, b0c.v, (short)0, acc10, false, false);
  acc11 = __builtin_amdgcn_wmma_f32_16x16x32_f16(false, a1c.v, false, b1c.v, (short)0, acc11, false, false);

  const float bv0 = bias[e * C_OUT + ncol0];
  const float bv1 = nt1_live ? bias[e * C_OUT + ncol0 + 16] : 0.0f;
  const int half = lane >> 4;                  // C/D layout: N=lane&15, M=r+8*half
#pragma unroll
  for (int r = 0; r < 8; ++r) {
    int mrow = r + half * 8;
    int dst0 = dstrow[rt0 * 16 + mrow];
    int dst1 = dstrow[rt0 * 16 + 16 + mrow];
    if (dst0 >= 0) {
      out[(size_t)dst0 * C_OUT + ncol0] = acc00[r] + bv0;
      if (nt1_live) out[(size_t)dst0 * C_OUT + ncol0 + 16] = acc01[r] + bv1;
    }
    if (dst1 >= 0) {
      out[(size_t)dst1 * C_OUT + ncol0] = acc10[r] + bv0;
      if (nt1_live) out[(size_t)dst1 * C_OUT + ncol0 + 16] = acc11[r] + bv1;
    }
  }
}

// ---------------- launch ----------------
extern "C" void kernel_launch(void* const* d_in, const int* in_sizes, int n_in,
                              void* d_out, int out_size, void* d_ws, size_t ws_size,
                              hipStream_t stream) {
  const float* x    = (const float*)d_in[0];   // [4096,256,3,3]
  const int*   aidx = (const int*)d_in[1];     // [4096]
  const float* W    = (const float*)d_in[2];   // [9,784,256,3,3]
  const float* b    = (const float*)d_in[3];   // [9,784]
  float* out = (float*)d_out;                  // [4096,784]

  // workspace layout
  char* ws = (char*)d_ws;
  int* perm       = (int*)ws;                  //  4096
  int* sortpos    = perm + NROWS;              //  4096
  int* srcrow     = sortpos + NROWS;           //  4384
  int* dstrow     = srcrow + CAP;              //  4384
  int* tileExpert = dstrow + CAP;              //   274
  const size_t WH_OFF = 131072;                // 128 KB, aligned
  const size_t WH_BYTES = (size_t)NUM_ANCHORS * C_OUT * KDIM * sizeof(_Float16);
  const size_t XS_OFF = WH_OFF + WH_BYTES + 131072;  // 128 KB slack: n-tile overhang + prefetch
  _Float16* Wh = (_Float16*)(ws + WH_OFF);     // 32.5 MB
  _Float16* Xs = (_Float16*)(ws + XS_OFF);     // 20.2 MB  (total ~53 MB)

  routing_kernel<<<1, 256, 0, stream>>>(aidx, perm, sortpos, srcrow, dstrow, tileExpert);

  const int nW = NUM_ANCHORS * C_OUT * KDIM;
  cvtW_kernel<<<(nW + 255) / 256, 256, 0, stream>>>(W, Wh, nW);

  gatherX_kernel<<<CAP, 256, 0, stream>>>(x, srcrow, Xs);

  dim3 grid(ROW_PAIRS, (NPAIRS + 3) / 4);
  moe_gemm_kernel<<<grid, 128, 0, stream>>>(Xs, Wh, b, tileExpert, dstrow, out);
}
